// GNN_DDAS_78469052498542
// MI455X (gfx1250) — matrix operand order repaired and checked by reading, not compiled, not run on hardware
//
#include <hip/hip_runtime.h>
#include <hip/hip_bf16.h>
#include <math.h>

typedef __bf16 bf16;
typedef __attribute__((ext_vector_type(16))) __bf16 v16bf;
typedef __attribute__((ext_vector_type(8)))  float  v8f;

// ---------------------------------------------------------------------------
// CDNA5 WMMA fragment layout helpers (ISA 7.12.2, wave32)
// A (16x32 bf16): lane g=l>>4, m=l&15 ; elem e -> VGPR v=e>>1, half h=e&1
//   k = (v>=4?16:0) + g*8 + (v&3)*2 + h
// B (32x16 bf16): n = l&15 ; k = (v>=4?16:0) + g*8 + (v&3)*2 + h
// C/D (16x16 f32): n = l&15 ; m = i + 8*g  (i = acc element 0..7)
// ---------------------------------------------------------------------------
__device__ __forceinline__ int a_k_of(int lane, int e) {
  int g = lane >> 4, v = e >> 1, h = e & 1;
  return ((v >= 4) ? 16 : 0) + g * 8 + (v & 3) * 2 + h;
}
__device__ __forceinline__ int b_k_of(int lane, int e) {
  int g = lane >> 4, v = e >> 1, h = e & 1;
  return ((v >= 4) ? 16 : 0) + g * 8 + (v & 3) * 2 + h;
}
// inverse of A layout: (row r 0..15, k' 0..31) -> (lane, elem)
__device__ __forceinline__ void a_pos_of(int r, int kk, int& lane, int& e) {
  int top  = (kk >= 16) ? 4 : 0;
  int rem  = kk & 15;
  int g    = rem >> 3;
  int rem2 = rem & 7;
  int v    = top + (rem2 >> 1);
  int h    = rem2 & 1;
  lane = g * 16 + r;
  e    = v * 2 + h;
}

__device__ __forceinline__ unsigned f2u_ord(float f) {
  unsigned u = __float_as_uint(f);
  return (u & 0x80000000u) ? ~u : (u | 0x80000000u);
}
__device__ __forceinline__ float u2f_ord(unsigned u) {
  return (u & 0x80000000u) ? __uint_as_float(u & 0x7fffffffu) : __uint_as_float(~u);
}
__device__ __forceinline__ void edge_sd(const int* ei, long long e, int E, int& s, int& d) {
  if (e < (long long)E) { s = ei[e]; d = ei[(size_t)E + e]; }
  else                  { s = d = (int)(e - E); }
}

// ---------------------------------------------------------------------------
// Packing kernels: f32 row-major -> bf16 fragment-packed (zero padded)
// A packed: [mtile][kfrag][512]   B packed: [ntile][kfrag][512]
// ---------------------------------------------------------------------------
__global__ void k_pack_a(const float* __restrict__ src, bf16* __restrict__ dst,
                         int M, int K, int mtiles, int kfrags) {
  long long total = (long long)mtiles * kfrags * 512;
  for (long long i = (long long)blockIdx.x * 256 + threadIdx.x; i < total;
       i += (long long)gridDim.x * 256) {
    int le = (int)(i & 511);
    long long t = i >> 9;
    int kf = (int)(t % kfrags);
    int mt = (int)(t / kfrags);
    int lane = le >> 4, e = le & 15;
    int m = mt * 16 + (lane & 15);
    int k = kf * 32 + a_k_of(lane, e);
    float v = (m < M && k < K) ? src[(size_t)m * K + k] : 0.f;
    dst[i] = (bf16)v;
  }
}

__global__ void k_pack_b(const float* __restrict__ src, bf16* __restrict__ dst,
                         int K, int Nc, int ntiles, int kfrags) {
  long long total = (long long)ntiles * kfrags * 512;
  for (long long i = (long long)blockIdx.x * 256 + threadIdx.x; i < total;
       i += (long long)gridDim.x * 256) {
    int le = (int)(i & 511);
    long long t = i >> 9;
    int kf = (int)(t % kfrags);
    int nt = (int)(t / kfrags);
    int lane = le >> 4, e = le & 15;
    int n = nt * 16 + (lane & 15);
    int k = kf * 32 + b_k_of(lane, e);
    float v = (k < K && n < Nc) ? src[(size_t)k * Nc + n] : 0.f;
    dst[i] = (bf16)v;
  }
}

// ---------------------------------------------------------------------------
// Fragment-packed WMMA GEMM: C[M,Nc] = A @ B (+bias)(+relu), f32 out
// block = 128 (4 waves), wave w -> mtile = blockIdx.x*4+w, ntile = blockIdx.y
// ---------------------------------------------------------------------------
__global__ __launch_bounds__(128) void k_gemm(
    const bf16* __restrict__ Apk, const bf16* __restrict__ Bpk,
    const float* __restrict__ bias, float* __restrict__ C,
    int M, int Nc, int mtiles, int kfrags, int relu) {
  int lane = threadIdx.x & 31, w = threadIdx.x >> 5;
  int mtile = blockIdx.x * 4 + w;
  int ntile = blockIdx.y;
  if (mtile >= mtiles) return;
  v8f acc = {};
  const bf16* ap = Apk + (size_t)mtile * kfrags * 512 + lane * 16;
  const bf16* bp = Bpk + (size_t)ntile * kfrags * 512 + lane * 16;
  for (int kf = 0; kf < kfrags; ++kf) {
    v16bf a = *(const v16bf*)(ap + (size_t)kf * 512);
    v16bf b = *(const v16bf*)(bp + (size_t)kf * 512);
    acc = __builtin_amdgcn_wmma_f32_16x16x32_bf16(false, a, false, b, (short)0,
                                                  acc, false, false);
  }
  int g = lane >> 4;
  int n = ntile * 16 + (lane & 15);
  float bv = bias ? bias[n] : 0.f;
#pragma unroll
  for (int i = 0; i < 8; ++i) {
    int m = mtile * 16 + i + 8 * g;
    if (m < M) {
      float v = acc[i] + bv;
      if (relu) v = fmaxf(v, 0.f);
      C[(size_t)m * Nc + n] = v;
    }
  }
}

// ---------------------------------------------------------------------------
// GAT attention pieces
// ---------------------------------------------------------------------------
__global__ void k_scores(const float* __restrict__ h, const float* __restrict__ asrc,
                         const float* __restrict__ adst, float* __restrict__ ssrc,
                         float* __restrict__ sdst, int N) {
  int idx = blockIdx.x * 256 + threadIdx.x;
  if (idx >= N * 4) return;
  int n = idx >> 2, hd = idx & 3;
  const float* hp = h + (size_t)n * 128 + hd * 32;
  float s0 = 0.f, s1 = 0.f;
  for (int c = 0; c < 32; ++c) {
    float v = hp[c];
    s0 += v * asrc[hd * 32 + c];
    s1 += v * adst[hd * 32 + c];
  }
  ssrc[idx] = s0;
  sdst[idx] = s1;
}

__global__ void k_fill_f32(float* p, float v, long long n) {
  long long i = (long long)blockIdx.x * 256 + threadIdx.x;
  if (i < n) p[i] = v;
}
__global__ void k_fill_u32(unsigned* p, unsigned v, long long n) {
  long long i = (long long)blockIdx.x * 256 + threadIdx.x;
  if (i < n) p[i] = v;
}

__global__ void k_edge_max(const int* __restrict__ ei, const float* __restrict__ ssrc,
                           const float* __restrict__ sdst, unsigned* __restrict__ mmax,
                           int E, long long Etot) {
  long long idx = (long long)blockIdx.x * 256 + threadIdx.x;
  if (idx >= Etot * 4) return;
  long long e = idx >> 2;
  int hd = (int)(idx & 3);
  int s, d;
  edge_sd(ei, e, E, s, d);
  float ev = ssrc[(size_t)s * 4 + hd] + sdst[(size_t)d * 4 + hd];
  ev = ev >= 0.f ? ev : 0.2f * ev;
  atomicMax(mmax + (size_t)d * 4 + hd, f2u_ord(ev));
}

__global__ void k_edge_exp(const int* __restrict__ ei, const float* __restrict__ ssrc,
                           const float* __restrict__ sdst, const unsigned* __restrict__ mmax,
                           float* __restrict__ den, float* __restrict__ exbuf,
                           int E, long long Etot) {
  long long idx = (long long)blockIdx.x * 256 + threadIdx.x;
  if (idx >= Etot * 4) return;
  long long e = idx >> 2;
  int hd = (int)(idx & 3);
  int s, d;
  edge_sd(ei, e, E, s, d);
  float ev = ssrc[(size_t)s * 4 + hd] + sdst[(size_t)d * 4 + hd];
  ev = ev >= 0.f ? ev : 0.2f * ev;
  float ex = expf(ev - u2f_ord(mmax[(size_t)d * 4 + hd]));
  exbuf[idx] = ex;
  atomicAdd(den + (size_t)d * 4 + hd, ex);
}

__global__ void k_edge_scatter(const int* __restrict__ ei, const float* __restrict__ exbuf,
                               const float* __restrict__ den, const float* __restrict__ h,
                               float* __restrict__ outacc, int E, long long Etot) {
  long long idx = (long long)blockIdx.x * 256 + threadIdx.x;
  if (idx >= Etot * 128) return;
  long long e = idx >> 7;
  int f = (int)(idx & 127);
  int hd = f >> 5;
  int s, d;
  edge_sd(ei, e, E, s, d);
  float alpha = exbuf[e * 4 + hd] / (den[(size_t)d * 4 + hd] + 1e-16f);
  atomicAdd(outacc + (size_t)d * 128 + f, h[(size_t)s * 128 + f] * alpha);
}

// LayerNorm over 128 features, one wave per node, optional leaky(0.01)
__global__ __launch_bounds__(256) void k_ln(
    const float* __restrict__ in, const float* __restrict__ bias,
    const float* __restrict__ g, const float* __restrict__ beta,
    float* __restrict__ out, int N, int leaky) {
  int lane = threadIdx.x & 31, w = threadIdx.x >> 5;
  int n = blockIdx.x * 8 + w;
  if (n >= N) return;
  float v[4], s = 0.f, s2 = 0.f;
#pragma unroll
  for (int j = 0; j < 4; ++j) {
    int f = lane + j * 32;
    v[j] = in[(size_t)n * 128 + f] + bias[f];
    s += v[j];
    s2 += v[j] * v[j];
  }
  for (int off = 16; off >= 1; off >>= 1) {
    s  += __shfl_xor(s,  off, 32);
    s2 += __shfl_xor(s2, off, 32);
  }
  float mu  = s * (1.f / 128.f);
  float var = s2 * (1.f / 128.f) - mu * mu;
  float inv = rsqrtf(var + 1e-5f);
#pragma unroll
  for (int j = 0; j < 4; ++j) {
    int f = lane + j * 32;
    float o = (v[j] - mu) * inv * g[f] + beta[f];
    if (leaky) o = o >= 0.f ? o : 0.01f * o;
    out[(size_t)n * 128 + f] = o;
  }
}

__global__ void k_pool_add(const float* __restrict__ h, const int* __restrict__ batch,
                           float* __restrict__ pooled, float* __restrict__ counts, int N) {
  long long idx = (long long)blockIdx.x * 256 + threadIdx.x;
  if (idx >= (long long)N * 128) return;
  int n = (int)(idx >> 7), f = (int)(idx & 127);
  int b = batch[n];
  atomicAdd(pooled + (size_t)b * 128 + f, h[idx]);
  if (f == 0) atomicAdd(counts + b, 1.f);
}
__global__ void k_pool_div(float* pooled, const float* counts, int B) {
  int idx = blockIdx.x * 256 + threadIdx.x;
  if (idx >= B * 128) return;
  pooled[idx] /= fmaxf(counts[idx >> 7], 1.f);
}

// ---------------------------------------------------------------------------
// Fused one-hot branch: one WG (256 thr, 8 waves) per molecule.
// LDS-staged WMMA: one_hot[b] (128x65) -> fc1(+relu) -> fc2 -> column mean.
// ---------------------------------------------------------------------------
__global__ __launch_bounds__(256) void k_onehot(
    const float* __restrict__ onehot, const bf16* __restrict__ w1pk,
    const float* __restrict__ b1, const bf16* __restrict__ w2pk,
    const float* __restrict__ b2, float* __restrict__ o256) {
  __shared__ __align__(32) bf16 ldsA[8 * 3 * 512];  // one_hot A-frags (24KB)
  __shared__ __align__(32) bf16 ldsO[8 * 4 * 512];  // fc1 out A-frags (32KB)
  __shared__ float colsum[256];
  int b = blockIdx.x;
  int tid = threadIdx.x, lane = tid & 31, w = tid >> 5;
  colsum[tid] = 0.f;
  for (int i = tid; i < 8 * 3 * 512; i += 256) {
    int le = i & 511, t = i >> 9;
    int kf = t % 3, mt = t / 3;
    int ln = le >> 4, e = le & 15;
    int m = mt * 16 + (ln & 15);
    int k = kf * 32 + a_k_of(ln, e);
    ldsA[i] = (bf16)((k < 65) ? onehot[((size_t)b * 128 + m) * 65 + k] : 0.f);
  }
  __syncthreads();
  int g = lane >> 4;
  // fc1: mtile = w, ntile = 0..7 ; K = 65 -> 3 kfrags
  for (int nt = 0; nt < 8; ++nt) {
    v8f acc = {};
#pragma unroll
    for (int kf = 0; kf < 3; ++kf) {
      v16bf a  = *(const v16bf*)(ldsA + (w * 3 + kf) * 512 + lane * 16);
      v16bf bb = *(const v16bf*)(w1pk + (size_t)(nt * 3 + kf) * 512 + lane * 16);
      acc = __builtin_amdgcn_wmma_f32_16x16x32_bf16(false, a, false, bb, (short)0,
                                                    acc, false, false);
    }
    int col = nt * 16 + (lane & 15);
    float bv = b1[col];
    int kf2 = col >> 5, kk = col & 31;
#pragma unroll
    for (int i = 0; i < 8; ++i) {
      int r = i + 8 * g;  // row within tile (mtile = w)
      float val = fmaxf(acc[i] + bv, 0.f);
      int l2, e2;
      a_pos_of(r, kk, l2, e2);
      ldsO[(w * 4 + kf2) * 512 + l2 * 16 + e2] = (bf16)val;
    }
  }
  __syncthreads();
  // fc2: mtile = w, ntile = 0..15 ; K = 128 -> 4 kfrags; column-sum accumulate
  for (int nt = 0; nt < 16; ++nt) {
    v8f acc = {};
#pragma unroll
    for (int kf = 0; kf < 4; ++kf) {
      v16bf a  = *(const v16bf*)(ldsO + (w * 4 + kf) * 512 + lane * 16);
      v16bf bb = *(const v16bf*)(w2pk + (size_t)(nt * 4 + kf) * 512 + lane * 16);
      acc = __builtin_amdgcn_wmma_f32_16x16x32_bf16(false, a, false, bb, (short)0,
                                                    acc, false, false);
    }
    int col = nt * 16 + (lane & 15);
    float csum = 0.f;
#pragma unroll
    for (int i = 0; i < 8; ++i) csum += acc[i];
    atomicAdd(&colsum[col], csum);
  }
  __syncthreads();
  o256[(size_t)b * 256 + tid] = colsum[tid] * (1.f / 128.f) + b2[tid];
}

// token branch: mean over L of embedding gathers
__global__ void k_tok(const int* __restrict__ toks, const float* __restrict__ emb,
                      float* __restrict__ temb, int B) {
  int idx = blockIdx.x * 256 + threadIdx.x;
  if (idx >= B * 64) return;
  int b = idx >> 6, c = idx & 63;
  float s = 0.f;
  for (int l = 0; l < 128; ++l) s += emb[(size_t)toks[b * 128 + l] * 64 + c];
  temb[idx] = s * (1.f / 128.f);
}

__global__ void k_concat(const float* __restrict__ ge, const float* __restrict__ oe,
                         const float* __restrict__ te, float* __restrict__ feat, int B) {
  int idx = blockIdx.x * 256 + threadIdx.x;
  if (idx >= B * 256) return;
  int b = idx >> 8, j = idx & 255;
  feat[idx] = (j < 128) ? ge[(size_t)b * 128 + j]
                        : (oe[(size_t)b * 128 + j - 128] + te[(size_t)b * 128 + j - 128]);
}

__global__ void k_head4(const float* __restrict__ f3, const float* __restrict__ w4,
                        const float* __restrict__ b4, float* __restrict__ out, int B) {
  int b = blockIdx.x * 256 + threadIdx.x;
  if (b >= B) return;
  float s = b4[0];
  for (int j = 0; j < 32; ++j) s += f3[(size_t)b * 32 + j] * w4[j];
  out[b] = 1.f / (1.f + expf(-s));
}

// ---------------------------------------------------------------------------
extern "C" void kernel_launch(void* const* d_in, const int* in_sizes, int n_in,
                              void* d_out, int out_size, void* d_ws, size_t ws_size,
                              hipStream_t stream) {
  const float* x      = (const float*)d_in[0];
  const int*   ei     = (const int*)d_in[1];
  const int*   toks   = (const int*)d_in[2];
  const float* onehot = (const float*)d_in[3];
  const int*   batch  = (const int*)d_in[5];

  const int N = in_sizes[0] / 78;
  const int E = in_sizes[1] / 2;
  const int B = in_sizes[2] / 128;
  const long long Etot = (long long)E + N;

  int pi = 6;
  struct GatP { const float *W, *asrc, *adst, *b, *g, *beta; } gp[3];
  for (int l = 0; l < 3; ++l) {
    gp[l].W    = (const float*)d_in[pi++];
    gp[l].asrc = (const float*)d_in[pi++];
    gp[l].adst = (const float*)d_in[pi++];
    gp[l].b    = (const float*)d_in[pi++];
    gp[l].g    = (const float*)d_in[pi++];
    gp[l].beta = (const float*)d_in[pi++];
  }
  const float* flat_w = (const float*)d_in[pi++]; const float* flat_b = (const float*)d_in[pi++];
  const float* fc1_w  = (const float*)d_in[pi++]; const float* fc1_b  = (const float*)d_in[pi++];
  const float* fc2_w  = (const float*)d_in[pi++]; const float* fc2_b  = (const float*)d_in[pi++];
  const float* emb    = (const float*)d_in[pi++];
  const float* toks_w = (const float*)d_in[pi++]; const float* toks_b = (const float*)d_in[pi++];
  const float* one_w  = (const float*)d_in[pi++]; const float* one_b  = (const float*)d_in[pi++];
  const float* h1_w   = (const float*)d_in[pi++]; const float* h1_b   = (const float*)d_in[pi++];
  const float* h2_w   = (const float*)d_in[pi++]; const float* h2_b   = (const float*)d_in[pi++];
  const float* h3_w   = (const float*)d_in[pi++]; const float* h3_b   = (const float*)d_in[pi++];
  const float* h4_w   = (const float*)d_in[pi++]; const float* h4_b   = (const float*)d_in[pi++];

  auto cdiv = [](long long a, long long b) -> long long { return (a + b - 1) / b; };

  // ---- workspace carve-out ----
  unsigned char* base = (unsigned char*)d_ws;
  size_t off = 0;
  auto alloc = [&](size_t bytes) -> void* {
    void* p = base + off;
    off += (bytes + 255) & ~(size_t)255;
    return p;
  };
  const int mtilesN = (int)cdiv(N, 16);
  float*    hlin   = (float*)alloc((size_t)N * 128 * 4);
  float*    hacc   = (float*)alloc((size_t)N * 128 * 4);
  float*    hln    = (float*)alloc((size_t)N * 128 * 4);
  bf16*     apackN = (bf16*)alloc((size_t)mtilesN * 4 * 512 * 2);
  float*    ssrc   = (float*)alloc((size_t)N * 4 * 4);
  float*    sdst   = (float*)alloc((size_t)N * 4 * 4);
  unsigned* mmax   = (unsigned*)alloc((size_t)N * 4 * 4);
  float*    den    = (float*)alloc((size_t)N * 4 * 4);
  float*    exbuf  = (float*)alloc((size_t)Etot * 4 * 4);
  float*    pooled = (float*)alloc((size_t)B * 128 * 4);
  float*    counts = (float*)alloc((size_t)B * 4);
  float*    gemb   = (float*)alloc((size_t)B * 128 * 4);
  float*    o256   = (float*)alloc((size_t)B * 256 * 4);
  float*    temb   = (float*)alloc((size_t)B * 64 * 4);
  float*    tokemb = (float*)alloc((size_t)B * 128 * 4);
  float*    oneemb = (float*)alloc((size_t)B * 128 * 4);
  float*    feat   = (float*)alloc((size_t)B * 256 * 4);
  float*    f1     = (float*)alloc((size_t)B * 128 * 4);
  float*    f2     = (float*)alloc((size_t)B * 64 * 4);
  float*    f3     = (float*)alloc((size_t)B * 32 * 4);
  const int mtilesB = (int)cdiv(B, 16);
  bf16*     apackB  = (bf16*)alloc((size_t)mtilesB * 8 * 512 * 2);
  bf16* wpk[3];
  wpk[0]       = (bf16*)alloc((size_t)8 * 3 * 512 * 2);
  wpk[1]       = (bf16*)alloc((size_t)8 * 4 * 512 * 2);
  wpk[2]       = (bf16*)alloc((size_t)8 * 4 * 512 * 2);
  bf16* flatpk = (bf16*)alloc((size_t)8 * 4 * 512 * 2);
  bf16* fc1pk  = (bf16*)alloc((size_t)8 * 3 * 512 * 2);
  bf16* fc2pk  = (bf16*)alloc((size_t)16 * 4 * 512 * 2);
  bf16* tokpk  = (bf16*)alloc((size_t)8 * 2 * 512 * 2);
  bf16* onepk  = (bf16*)alloc((size_t)8 * 8 * 512 * 2);
  bf16* h1pk   = (bf16*)alloc((size_t)8 * 8 * 512 * 2);
  bf16* h2pk   = (bf16*)alloc((size_t)4 * 4 * 512 * 2);
  bf16* h3pk   = (bf16*)alloc((size_t)2 * 2 * 512 * 2);
  (void)ws_size;

  auto packB = [&](const float* W, bf16* dst, int K, int Nc) {
    int nt = Nc / 16, kf = (int)cdiv(K, 32);
    long long tot = (long long)nt * kf * 512;
    k_pack_b<<<(unsigned)cdiv(tot, 256), 256, 0, stream>>>(W, dst, K, Nc, nt, kf);
  };
  auto packA = [&](const float* A, bf16* dst, long long M, int K) {
    int mt = (int)cdiv(M, 16), kf = (int)cdiv(K, 32);
    long long tot = (long long)mt * kf * 512;
    k_pack_a<<<(unsigned)cdiv(tot, 256), 256, 0, stream>>>(A, dst, (int)M, K, mt, kf);
  };
  auto gemm = [&](const bf16* A, const bf16* Bp, const float* bias, float* C,
                  long long M, int Nc, int K, int relu) {
    int mt = (int)cdiv(M, 16), kf = (int)cdiv(K, 32);
    dim3 gr((unsigned)cdiv(mt, 4), Nc / 16);
    k_gemm<<<gr, 128, 0, stream>>>(A, Bp, bias, C, (int)M, Nc, mt, kf, relu);
  };

  // ---- pack weights ----
  packB(gp[0].W, wpk[0], 78, 128);
  packB(gp[1].W, wpk[1], 128, 128);
  packB(gp[2].W, wpk[2], 128, 128);
  packB(flat_w, flatpk, 128, 128);
  packB(fc1_w, fc1pk, 65, 128);
  packB(fc2_w, fc2pk, 128, 256);
  packB(toks_w, tokpk, 64, 128);
  packB(one_w, onepk, 256, 128);
  packB(h1_w, h1pk, 256, 128);
  packB(h2_w, h2pk, 128, 64);
  packB(h3_w, h3pk, 64, 32);

  // ---- GAT layers ----
  const float* act = x;
  int K = 78;
  for (int l = 0; l < 3; ++l) {
    packA(act, apackN, N, K);
    gemm(apackN, wpk[l], nullptr, hlin, N, 128, K, 0);
    k_scores<<<(unsigned)cdiv((long long)N * 4, 256), 256, 0, stream>>>(
        hlin, gp[l].asrc, gp[l].adst, ssrc, sdst, N);
    k_fill_u32<<<(unsigned)cdiv((long long)N * 4, 256), 256, 0, stream>>>(mmax, 0u, (long long)N * 4);
    k_fill_f32<<<(unsigned)cdiv((long long)N * 4, 256), 256, 0, stream>>>(den, 0.f, (long long)N * 4);
    k_fill_f32<<<(unsigned)cdiv((long long)N * 128, 256), 256, 0, stream>>>(hacc, 0.f, (long long)N * 128);
    k_edge_max<<<(unsigned)cdiv(Etot * 4, 256), 256, 0, stream>>>(ei, ssrc, sdst, mmax, E, Etot);
    k_edge_exp<<<(unsigned)cdiv(Etot * 4, 256), 256, 0, stream>>>(ei, ssrc, sdst, mmax, den, exbuf, E, Etot);
    k_edge_scatter<<<(unsigned)cdiv(Etot * 128, 256), 256, 0, stream>>>(ei, exbuf, den, hlin, hacc, E, Etot);
    k_ln<<<(unsigned)cdiv(N, 8), 256, 0, stream>>>(hacc, gp[l].b, gp[l].g, gp[l].beta, hln, N, l < 2 ? 1 : 0);
    act = hln;
    K = 128;
  }

  // ---- global mean pool + graph embedding ----
  k_fill_f32<<<(unsigned)cdiv((long long)B * 128, 256), 256, 0, stream>>>(pooled, 0.f, (long long)B * 128);
  k_fill_f32<<<(unsigned)cdiv(B, 256), 256, 0, stream>>>(counts, 0.f, B);
  k_pool_add<<<(unsigned)cdiv((long long)N * 128, 256), 256, 0, stream>>>(hln, batch, pooled, counts, N);
  k_pool_div<<<(unsigned)cdiv((long long)B * 128, 256), 256, 0, stream>>>(pooled, counts, B);
  packA(pooled, apackB, B, 128);
  gemm(apackB, flatpk, flat_b, gemb, B, 128, 128, 0);

  // ---- one-hot branch (fused) + projection ----
  k_onehot<<<B, 256, 0, stream>>>(onehot, fc1pk, fc1_b, fc2pk, fc2_b, o256);
  packA(o256, apackB, B, 256);
  gemm(apackB, onepk, one_b, oneemb, B, 128, 256, 0);

  // ---- token branch ----
  k_tok<<<(unsigned)cdiv((long long)B * 64, 256), 256, 0, stream>>>(toks, emb, temb, B);
  packA(temb, apackB, B, 64);
  gemm(apackB, tokpk, toks_b, tokemb, B, 128, 64, 0);

  // ---- head MLP ----
  k_concat<<<(unsigned)cdiv((long long)B * 256, 256), 256, 0, stream>>>(gemb, oneemb, tokemb, feat, B);
  packA(feat, apackB, B, 256);
  gemm(apackB, h1pk, h1_b, f1, B, 128, 256, 0);
  packA(f1, apackB, B, 128);
  gemm(apackB, h2pk, h2_b, f2, B, 64, 128, 0);
  packA(f2, apackB, B, 64);
  gemm(apackB, h3pk, h3_b, f3, B, 32, 64, 1);
  k_head4<<<(unsigned)cdiv(B, 256), 256, 0, stream>>>(f3, h4_w, h4_b, (float*)d_out, B);
  (void)n_in; (void)out_size;
}
